// EnergyConditionedEquivariantAtomConvolution_52905407152570
// MI455X (gfx1250) — matrix operand order repaired and checked by reading, not compiled, not run on hardware
//
#include <hip/hip_runtime.h>
#include <math.h>

// ---------------------------------------------------------------------------
// Problem constants (from reference)
// ---------------------------------------------------------------------------
#define BB      64
#define NN      128
#define HDIM    128
#define NODE_S  64
#define NODE_V  32
#define OUT_S   32
#define OUT_V   16
#define RBF     16
#define ZEMB    32
#define HID     128
#define LAT     256
#define EDIM    16
#define NE      512
#define EATT    4096
#define FLATN   (BB * NN)          // 8192
#define WN      4608               // 64*32 + 32*32 + 64*16 + 32*16
#define NODE_DIM 160
#define OUT_DIM  80
#define CUTOFF  5.0f

typedef __attribute__((ext_vector_type(16))) _Float16 f16x16;
typedef __attribute__((ext_vector_type(8)))  _Float16 f16x8;
typedef __attribute__((ext_vector_type(8)))  float    f32x8;

__device__ __forceinline__ float silu_f(float x) { return x / (1.0f + expf(-x)); }

// ---------------------------------------------------------------------------
// WMMA helpers (gfx1250 wave32, V_WMMA_F32_16X16X32_F16)
// A 16x32 f16: lane m=l&15, kb=(l<16)?0:8; elems = K[k0+kb .. +7] ++ K[k0+kb+16 .. +23]
//   -> two contiguous 8-half runs: vectorized 16B LDS loads.
// B 32x16 f16: lane n=l&15, kb=(l<16)?0:8; elem j -> K=kb+(j&7)+(j>=8?16:0), as A.  Pre-packed in
//   global memory as [tile][lane][16] so a fragment is one 32B per-lane load.
// C/D 16x16 f32: vgpr r, lane l: M = r + ((l<16)?0:8), N = l&15
// ---------------------------------------------------------------------------
__device__ __forceinline__ f16x16 frag_a_lds(const _Float16* s, int ld, int k0) {
  int l = threadIdx.x & 31;
  const _Float16* p = s + (l & 15) * ld + k0 + ((l < 16) ? 0 : 8);
  f16x8 lo = *(const f16x8*)p;
  f16x8 hi = *(const f16x8*)(p + 16);
  return __builtin_shufflevector(lo, hi, 0, 1, 2, 3, 4, 5, 6, 7,
                                 8, 9, 10, 11, 12, 13, 14, 15);
}

// Packed-B fragment: ktiles = Kp/32, kt = k0/32, nt = n0/16.
__device__ __forceinline__ f16x16 frag_b_pk(const _Float16* pk, int ktiles,
                                            int kt, int nt) {
  int l = threadIdx.x & 31;
  return *((const f16x16*)pk + (nt * ktiles + kt) * 32 + l);
}

__device__ __forceinline__ f32x8 wmma16(f16x16 a, f16x16 b, f32x8 c) {
  return __builtin_amdgcn_wmma_f32_16x16x32_f16(false, a, false, b,
                                                (short)0, c, false, false);
}

// ---------------------------------------------------------------------------
// K0: pack a row-major f32 weight [Ktrue x Np] into fragment-native f16 tiles
// (zero-padded to Kp). dst elem pe -> tile=pe>>9, lane=(pe>>4)&31, j=pe&15.
// ---------------------------------------------------------------------------
__global__ void k_pack(const float* __restrict__ src, _Float16* __restrict__ dst,
                       int Ktrue, int Kp, int Np) {
  int pe = blockIdx.x * blockDim.x + threadIdx.x;
  if (pe >= Kp * Np) return;
  int j = pe & 15;
  int lane = (pe >> 4) & 31;
  int tile = pe >> 9;
  int ktiles = Kp >> 5;
  int nt = tile / ktiles;
  int kt = tile - nt * ktiles;
  int n = nt * 16 + (lane & 15);
  int k = kt * 32 + ((lane < 16) ? 0 : 8) + (j & 7) + ((j >= 8) ? 16 : 0);
  float v = (k < Ktrue) ? src[k * Np + n] : 0.0f;
  dst[pe] = (_Float16)v;
}

// ---------------------------------------------------------------------------
// K1: scatter edges into flat per-row arrays
// ---------------------------------------------------------------------------
__global__ void k_scatter(const int* __restrict__ att_dst,
                          const float* __restrict__ att_dist,
                          const float* __restrict__ att_vec,
                          float* __restrict__ dflat,
                          float* __restrict__ vflat,
                          float* __restrict__ amask) {
  int e = blockIdx.x * blockDim.x + threadIdx.x;
  if (e >= EATT) return;
  int dst = att_dst[e];
  dflat[dst] = att_dist[e];
  vflat[dst * 3 + 0] = att_vec[e * 3 + 0];
  vflat[dst * 3 + 1] = att_vec[e * 3 + 1];
  vflat[dst * 3 + 2] = att_vec[e * 3 + 2];
  amask[dst] = 1.0f;
}

// ---------------------------------------------------------------------------
// K2: per-row features -> radial hidden (silu(win@w1_rad+b1), stored f16)
//     and gate = sigmoid(silu(gate_in@wg1+bg1)@wg2+bg2).
// One block = 16 flat rows (same batch), 128 threads = 4 waves.
// ---------------------------------------------------------------------------
__global__ __launch_bounds__(128) void k_rowfeat(
    const float* __restrict__ h, const int* __restrict__ z,
    const int* __restrict__ absorber, const float* __restrict__ w_zemb,
    const _Float16* __restrict__ pk_w1, const float* __restrict__ b1_rad,
    const _Float16* __restrict__ pk_wg1, const float* __restrict__ bg1,
    const float* __restrict__ wg2, const float* __restrict__ bg2,
    const float* __restrict__ dflat,
    _Float16* __restrict__ hidR, float* __restrict__ gate) {
  const float WIDTH = CUTOFF / (RBF - 1);
  __shared__ __align__(16) _Float16 sWin[16 * 64];
  __shared__ __align__(16) _Float16 sGin[16 * 288];
  __shared__ __align__(16) float    sHg[16 * 128];

  int tid = threadIdx.x;
  int wv = tid >> 5;
  int rows0 = blockIdx.x * 16;
  int b = rows0 >> 7;
  int absn = absorber[b];

  // build win tile: [zr(32)|isabs(1)|rbf(16)|pad] -> K=64
  for (int i = tid; i < 16 * 64; i += 128) {
    int r = i >> 6, c = i & 63;
    int row = rows0 + r;
    float v = 0.0f;
    if (c < 32) v = w_zemb[z[row] * ZEMB + c];
    else if (c == 32) v = ((row & 127) == absn) ? 1.0f : 0.0f;
    else if (c < 49) {
      int j = c - 33;
      float d = dflat[row];
      float t = (d - (float)j * WIDTH) / WIDTH;
      v = expf(-0.5f * t * t);
    }
    sWin[i] = (_Float16)v;
  }
  // build gate_in tile: [h_abs(128)|h(128)|rbf(16)|isabs|pad] -> K=288
  for (int i = tid; i < 16 * 288; i += 128) {
    int r = i / 288, c = i % 288;
    int row = rows0 + r;
    float v = 0.0f;
    if (c < 128) v = h[(b * NN + absn) * HDIM + c];
    else if (c < 256) v = h[row * HDIM + (c - 128)];
    else if (c < 272) {
      int j = c - 256;
      float d = dflat[row];
      float t = (d - (float)j * WIDTH) / WIDTH;
      v = expf(-0.5f * t * t);
    } else if (c == 272) v = ((row & 127) == absn) ? 1.0f : 0.0f;
    sGin[i] = (_Float16)v;
  }
  __syncthreads();

  int l = tid & 31;
  int mb = (l < 16) ? 0 : 8;
  int nloc = l & 15;

  // radial hidden: (16x64)@(64x128); wave wv covers cols [wv*32, wv*32+32)
  {
    f32x8 c0 = {}, c1 = {};
    int ntb = wv * 2;
#pragma unroll
    for (int kt = 0; kt < 2; ++kt) {
      f16x16 a = frag_a_lds(sWin, 64, kt * 32);
      c0 = wmma16(a, frag_b_pk(pk_w1, 2, kt, ntb + 0), c0);
      c1 = wmma16(a, frag_b_pk(pk_w1, 2, kt, ntb + 1), c1);
    }
#pragma unroll
    for (int q = 0; q < 2; ++q) {
      const f32x8& cc = q ? c1 : c0;
      int col = (ntb + q) * 16 + nloc;
      float bias = b1_rad[col];
#pragma unroll
      for (int r = 0; r < 8; ++r)
        hidR[(rows0 + r + mb) * HID + col] = (_Float16)silu_f(cc[r] + bias);
    }
  }

  // gate hidden: (16x288)@(288x128)
  {
    f32x8 c0 = {}, c1 = {};
    int ntb = wv * 2;
#pragma unroll
    for (int kt = 0; kt < 9; ++kt) {
      f16x16 a = frag_a_lds(sGin, 288, kt * 32);
      c0 = wmma16(a, frag_b_pk(pk_wg1, 9, kt, ntb + 0), c0);
      c1 = wmma16(a, frag_b_pk(pk_wg1, 9, kt, ntb + 1), c1);
    }
#pragma unroll
    for (int q = 0; q < 2; ++q) {
      const f32x8& cc = q ? c1 : c0;
      int col = (ntb + q) * 16 + nloc;
      float bias = bg1[col];
#pragma unroll
      for (int r = 0; r < 8; ++r)
        sHg[(r + mb) * 128 + col] = silu_f(cc[r] + bias);
    }
  }
  __syncthreads();

  if (tid < 16) {
    float acc = bg2[0];
    for (int k = 0; k < HID; ++k) acc += sHg[tid * 128 + k] * wg2[k];
    gate[rows0 + tid] = 1.0f / (1.0f + expf(-acc));
  }
}

// ---------------------------------------------------------------------------
// K3: fused big GEMM (hidden@w2_rad, tpw never hits HBM) + per-edge
//     equivariant einsums + env*gate*mask + batch reduction into vabs.
// One block = 16 edges (same batch). 4608 cols in 18 tiles of 256; region
// boundaries (wA:0-2047, wB:-3071, wC:-4095, wD:-4607) align to tiles.
// ---------------------------------------------------------------------------
__global__ __launch_bounds__(128) void k_tpw(
    const float* __restrict__ h_full, const _Float16* __restrict__ pk_w2,
    const float* __restrict__ b2_rad, const float* __restrict__ dflat,
    const float* __restrict__ vflat, const float* __restrict__ amask,
    const float* __restrict__ gate, const _Float16* __restrict__ hidR,
    float* __restrict__ vabs) {
  __shared__ __align__(16) _Float16 sHid[16 * 128];
  __shared__ float sS1[16 * 64];
  __shared__ float sV1[16 * 96];    // [e][i][c]
  __shared__ float sP[16 * 32];     // v1 . yv
  __shared__ float sYv[16 * 3];
  __shared__ float sMul[16];        // env*gate*amask
  __shared__ float sAccS[16 * 32];
  __shared__ float sAccC[16 * 16];
  __shared__ float sAccD[16 * 48];  // [e][o][c]
  __shared__ __align__(16) float sT[16 * 256];  // staged tpw tile

  int tid = threadIdx.x;
  int wv = tid >> 5;
  int rows0 = blockIdx.x * 16;
  int b = rows0 >> 7;

  // vectorized copy of the f16 hidden tile (16x128 = 256 uint4)
  {
    const uint4* src = (const uint4*)(hidR + rows0 * HID);
    uint4* dst = (uint4*)sHid;
    for (int i = tid; i < 256; i += 128) dst[i] = src[i];
  }
  for (int i = tid; i < 16 * NODE_DIM; i += 128) {
    int r = i / NODE_DIM, c = i % NODE_DIM;
    float v = h_full[(rows0 + r) * NODE_DIM + c];
    if (c < NODE_S) sS1[r * 64 + c] = v;
    else            sV1[r * 96 + (c - NODE_S)] = v;
  }
  for (int i = tid; i < 16 * 32; i += 128) sAccS[i] = 0.0f;
  for (int i = tid; i < 16 * 16; i += 128) sAccC[i] = 0.0f;
  for (int i = tid; i < 16 * 48; i += 128) sAccD[i] = 0.0f;
  if (tid < 16) {
    int row = rows0 + tid;
    float d = dflat[row];
    float eps = fmaxf(d, 1e-8f);
    float env = (d < CUTOFF)
                  ? 0.5f * (cosf(3.14159265358979323846f * d / CUTOFF) + 1.0f)
                  : 0.0f;
    sMul[tid] = env * gate[row] * amask[row];
    for (int c = 0; c < 3; ++c)
      sYv[tid * 3 + c] = 1.7320508075688772f * vflat[row * 3 + c] / eps;
  }
  __syncthreads();
  for (int i = tid; i < 16 * 32; i += 128) {
    int r = i >> 5, ii = i & 31;
    sP[i] = sV1[r * 96 + ii * 3 + 0] * sYv[r * 3 + 0] +
            sV1[r * 96 + ii * 3 + 1] * sYv[r * 3 + 1] +
            sV1[r * 96 + ii * 3 + 2] * sYv[r * 3 + 2];
  }
  __syncthreads();

  // A fragments of the 16x128 hidden tile are invariant across all 18 column
  // tiles: hoist them into VGPRs for the whole kernel (4 x 8 VGPRs).
  f16x16 ah[4];
#pragma unroll
  for (int kt = 0; kt < 4; ++kt) ah[kt] = frag_a_lds(sHid, 128, kt * 32);

  int l = tid & 31;
  int mb = (l < 16) ? 0 : 8;
  int nloc = l & 15;
  const float inv_s3 = 0.5773502691896258f;

  for (int t = 0; t < 18; ++t) {
    int tile0 = t * 256;
    // prefetch next tile's packed weights (64 KB) while this tile computes
    if (t + 1 < 18)
      __builtin_prefetch(pk_w2 + (t + 1) * 32768 + tid * 256, 0, 1);

    // WMMA: 16x256 tpw tile; wave wv covers 64 cols (4 N-tiles)
    int ntb = (tile0 >> 4) + wv * 4;
    f32x8 c[4] = {};
#pragma unroll
    for (int kt = 0; kt < 4; ++kt) {
#pragma unroll
      for (int q = 0; q < 4; ++q)
        c[q] = wmma16(ah[kt], frag_b_pk(pk_w2, 4, kt, ntb + q), c[q]);
    }
#pragma unroll
    for (int q = 0; q < 4; ++q) {
      int lc = wv * 64 + q * 16 + nloc;
      float bias = b2_rad[tile0 + lc];
#pragma unroll
      for (int r = 0; r < 8; ++r)
        sT[(r + mb) * 256 + lc] = c[q][r] + bias;
    }
    __syncthreads();

    // Consume (region uniform per tile; owner-exclusive accumulators)
    if (t < 8) {                       // wA: out_s += s1 . wA
      int i0 = t * 8;
      for (int s = tid; s < 512; s += 128) {
        int e = s >> 5, o = s & 31;
        float sum = 0.0f;
#pragma unroll
        for (int il = 0; il < 8; ++il)
          sum += sT[e * 256 + il * 32 + o] * sS1[e * 64 + i0 + il];
        sAccS[e * 32 + o] += sum;
      }
    } else if (t < 12) {               // wB: out_s += (v1.yv) . wB / sqrt(3)
      int i0 = (t - 8) * 8;
      for (int s = tid; s < 512; s += 128) {
        int e = s >> 5, o = s & 31;
        float sum = 0.0f;
#pragma unroll
        for (int il = 0; il < 8; ++il)
          sum += sT[e * 256 + il * 32 + o] * sP[e * 32 + i0 + il];
        sAccS[e * 32 + o] += inv_s3 * sum;
      }
    } else if (t < 16) {               // wC: scalar part of out_v
      int i0 = (t - 12) * 16;
      for (int s = tid; s < 256; s += 128) {
        int e = s >> 4, o = s & 15;
        float sum = 0.0f;
#pragma unroll
        for (int il = 0; il < 16; ++il)
          sum += sT[e * 256 + il * 16 + o] * sS1[e * 64 + i0 + il];
        sAccC[e * 16 + o] += sum;
      }
    } else {                           // wD: vector channels of out_v
      int i0 = (t - 16) * 16;
      for (int s = tid; s < 256; s += 128) {
        int e = s >> 4, o = s & 15;
        float a0 = 0.0f, a1 = 0.0f, a2 = 0.0f;
#pragma unroll
        for (int il = 0; il < 16; ++il) {
          float tv = sT[e * 256 + il * 16 + o];
          int vb = e * 96 + (i0 + il) * 3;
          a0 += tv * sV1[vb + 0];
          a1 += tv * sV1[vb + 1];
          a2 += tv * sV1[vb + 2];
        }
        int ab = (e * 16 + o) * 3;
        sAccD[ab + 0] += a0;
        sAccD[ab + 1] += a1;
        sAccD[ab + 2] += a2;
      }
    }
    __syncthreads();
  }

  // ve = (out_s|out_v) * norm * env*gate*mask ; reduce 16 edges -> vabs[b]
  const float norm = 0.10206207261596575f;  // 1/sqrt(NODE_S+NODE_V)
  if (tid < OUT_DIM) {
    float total = 0.0f;
    for (int e = 0; e < 16; ++e) {
      float m = sMul[e] * norm;
      float v;
      if (tid < OUT_S) {
        v = sAccS[e * 32 + tid];
      } else {
        int k = tid - OUT_S;
        int o = k / 3, cc = k % 3;
        v = sAccC[e * 16 + o] * sYv[e * 3 + cc] + sAccD[(e * 16 + o) * 3 + cc];
      }
      total += v * m;
    }
    atomicAdd(&vabs[b * OUT_DIM + tid], total);
  }
}

// ---------------------------------------------------------------------------
// K4: scales = silu(e_feat@we1+be1)@we2+be2  (tiny; one block per NE row)
// ---------------------------------------------------------------------------
__global__ __launch_bounds__(128) void k_scales(
    const float* __restrict__ e_feat, const float* __restrict__ we1,
    const float* __restrict__ be1, const float* __restrict__ we2,
    const float* __restrict__ be2, float* __restrict__ scales) {
  __shared__ float sh[HID];
  int r = blockIdx.x;
  int tid = threadIdx.x;
  float acc = be1[tid];
  for (int k = 0; k < EDIM; ++k) acc += e_feat[r * EDIM + k] * we1[k * HID + tid];
  sh[tid] = silu_f(acc);
  __syncthreads();
  if (tid < OUT_S + OUT_V) {
    float o = be2[tid];
    for (int k = 0; k < HID; ++k)
      o += sh[k] * we2[k * (OUT_S + OUT_V) + tid];
    scales[r * (OUT_S + OUT_V) + tid] = o;
  }
}

// ---------------------------------------------------------------------------
// K5: final MLP 48->128->128->256 over 32768 (b,e) rows; `inv` built on the
// fly from vabs x scales. Fully fused WMMA chain per 16-row tile.
// ---------------------------------------------------------------------------
__global__ __launch_bounds__(128) void k_final(
    const float* __restrict__ vabs, const float* __restrict__ scales,
    const _Float16* __restrict__ pk_wo1, const float* __restrict__ bo1,
    const _Float16* __restrict__ pk_wo2, const float* __restrict__ bo2,
    const _Float16* __restrict__ pk_wo3, const float* __restrict__ bo3,
    float* __restrict__ out) {
  __shared__ __align__(16) _Float16 sInv[16 * 64];
  __shared__ __align__(16) _Float16 sX1[16 * 128];
  __shared__ __align__(16) _Float16 sX2[16 * 128];

  int tid = threadIdx.x;
  int wv = tid >> 5;
  int R0 = blockIdx.x * 16;

  for (int i = tid; i < 16 * 64; i += 128) {
    int r = i >> 6, c = i & 63;
    int Rg = R0 + r;
    int bb = Rg >> 9;        // / NE
    int e = Rg & (NE - 1);
    float v = 0.0f;
    if (c < OUT_S) {
      v = vabs[bb * OUT_DIM + c] * scales[e * (OUT_S + OUT_V) + c];
    } else if (c < OUT_S + OUT_V) {
      int o = c - OUT_S;
      float sc = scales[e * (OUT_S + OUT_V) + OUT_S + o];
      float acc = 1e-12f;
#pragma unroll
      for (int cc = 0; cc < 3; ++cc) {
        float t = vabs[bb * OUT_DIM + OUT_S + o * 3 + cc] * sc;
        acc += t * t;
      }
      v = sqrtf(acc);
    }
    sInv[i] = (_Float16)v;
  }
  __syncthreads();

  int l = tid & 31;
  int mb = (l < 16) ? 0 : 8;
  int nloc = l & 15;

  // x1 = silu(inv@wo1+bo1), true K=48 (padded 64)
  {
    f32x8 c0 = {}, c1 = {};
    int ntb = wv * 2;
#pragma unroll
    for (int kt = 0; kt < 2; ++kt) {
      f16x16 a = frag_a_lds(sInv, 64, kt * 32);
      c0 = wmma16(a, frag_b_pk(pk_wo1, 2, kt, ntb + 0), c0);
      c1 = wmma16(a, frag_b_pk(pk_wo1, 2, kt, ntb + 1), c1);
    }
#pragma unroll
    for (int q = 0; q < 2; ++q) {
      const f32x8& cc = q ? c1 : c0;
      int col = (ntb + q) * 16 + nloc;
      float bias = bo1[col];
#pragma unroll
      for (int r = 0; r < 8; ++r)
        sX1[(r + mb) * 128 + col] = (_Float16)silu_f(cc[r] + bias);
    }
  }
  __syncthreads();

  // x2 = silu(x1@wo2+bo2)
  {
    f32x8 c0 = {}, c1 = {};
    int ntb = wv * 2;
#pragma unroll
    for (int kt = 0; kt < 4; ++kt) {
      f16x16 a = frag_a_lds(sX1, 128, kt * 32);
      c0 = wmma16(a, frag_b_pk(pk_wo2, 4, kt, ntb + 0), c0);
      c1 = wmma16(a, frag_b_pk(pk_wo2, 4, kt, ntb + 1), c1);
    }
#pragma unroll
    for (int q = 0; q < 2; ++q) {
      const f32x8& cc = q ? c1 : c0;
      int col = (ntb + q) * 16 + nloc;
      float bias = bo2[col];
#pragma unroll
      for (int r = 0; r < 8; ++r)
        sX2[(r + mb) * 128 + col] = (_Float16)silu_f(cc[r] + bias);
    }
  }
  __syncthreads();

  // out = x2@wo3+bo3 (N=256: wave covers 64 cols)
  {
    int ntb = wv * 4;
    f32x8 c[4] = {};
#pragma unroll
    for (int kt = 0; kt < 4; ++kt) {
      f16x16 a = frag_a_lds(sX2, 128, kt * 32);
#pragma unroll
      for (int q = 0; q < 4; ++q)
        c[q] = wmma16(a, frag_b_pk(pk_wo3, 4, kt, ntb + q), c[q]);
    }
#pragma unroll
    for (int q = 0; q < 4; ++q) {
      int col = (ntb + q) * 16 + nloc;
      float bias = bo3[col];
#pragma unroll
      for (int r = 0; r < 8; ++r)
        out[(R0 + r + mb) * LAT + col] = c[q][r] + bias;
    }
  }
}

// ---------------------------------------------------------------------------
// Launch
// ---------------------------------------------------------------------------
extern "C" void kernel_launch(void* const* d_in, const int* in_sizes, int n_in,
                              void* d_out, int out_size, void* d_ws,
                              size_t ws_size, hipStream_t stream) {
  const float* h        = (const float*)d_in[0];
  const float* h_full   = (const float*)d_in[1];
  const int*   z        = (const int*)d_in[2];
  /* d_in[3] = mask (unused by the reference core) */
  const float* e_feat   = (const float*)d_in[4];
  const int*   absorber = (const int*)d_in[5];
  const int*   att_dst  = (const int*)d_in[6];
  const float* att_dist = (const float*)d_in[7];
  const float* att_vec  = (const float*)d_in[8];
  const float* w_zemb   = (const float*)d_in[9];
  const float* w1_rad   = (const float*)d_in[10];
  const float* b1_rad   = (const float*)d_in[11];
  const float* w2_rad   = (const float*)d_in[12];
  const float* b2_rad   = (const float*)d_in[13];
  const float* wg1      = (const float*)d_in[14];
  const float* bg1      = (const float*)d_in[15];
  const float* wg2      = (const float*)d_in[16];
  const float* bg2      = (const float*)d_in[17];
  const float* we1      = (const float*)d_in[18];
  const float* be1      = (const float*)d_in[19];
  const float* we2      = (const float*)d_in[20];
  const float* be2      = (const float*)d_in[21];
  const float* wo1      = (const float*)d_in[22];
  const float* bo1      = (const float*)d_in[23];
  const float* wo2      = (const float*)d_in[24];
  const float* bo2      = (const float*)d_in[25];
  const float* wo3      = (const float*)d_in[26];
  const float* bo3      = (const float*)d_in[27];

  // workspace layout (float units): zero-init region first, then scratch
  float* ws      = (float*)d_ws;
  float* dflat   = ws;                 // 8192
  float* vflat   = ws + 8192;          // 24576
  float* amask   = ws + 32768;         // 8192
  float* vabs    = ws + 40960;         // 64*80 = 5120
  float* gate    = ws + 46080;         // 8192
  float* scales  = ws + 54272;         // 512*48 = 24576
  _Float16* hidR = (_Float16*)(ws + 78848);      // 8192*128 f16 (2 MB)
  _Float16* pk   = (_Float16*)(ws + 603136);     // packed weights (f16)
  _Float16* pk_w1  = pk;                 //  64x128  =   8192
  _Float16* pk_wg1 = pk + 8192;          // 288x128  =  36864
  _Float16* pk_w2  = pk + 45056;         // 128x4608 = 589824
  _Float16* pk_wo1 = pk + 634880;        //  64x128  =   8192
  _Float16* pk_wo2 = pk + 643072;        // 128x128  =  16384
  _Float16* pk_wo3 = pk + 659456;        // 128x256  =  32768

  hipMemsetAsync(dflat, 0, (size_t)(8192 + 24576 + 8192 + 5120) * sizeof(float),
                 stream);

  // pack weights into fragment-native f16 layout (L2-resident, ~1.4 MB)
  k_pack<<<(64 * 128) / 256, 256, 0, stream>>>(w1_rad, pk_w1, 49, 64, 128);
  k_pack<<<(288 * 128) / 256, 256, 0, stream>>>(wg1, pk_wg1, 273, 288, 128);
  k_pack<<<(128 * 4608) / 256, 256, 0, stream>>>(w2_rad, pk_w2, 128, 128, 4608);
  k_pack<<<(64 * 128) / 256, 256, 0, stream>>>(wo1, pk_wo1, 48, 64, 128);
  k_pack<<<(128 * 128) / 256, 256, 0, stream>>>(wo2, pk_wo2, 128, 128, 128);
  k_pack<<<(128 * 256) / 256, 256, 0, stream>>>(wo3, pk_wo3, 128, 128, 256);

  k_scatter<<<EATT / 256, 256, 0, stream>>>(att_dst, att_dist, att_vec,
                                            dflat, vflat, amask);
  k_rowfeat<<<FLATN / 16, 128, 0, stream>>>(h, z, absorber, w_zemb, pk_w1,
                                            b1_rad, pk_wg1, bg1, wg2, bg2,
                                            dflat, hidR, gate);
  k_tpw<<<FLATN / 16, 128, 0, stream>>>(h_full, pk_w2, b2_rad, dflat, vflat,
                                        amask, gate, hidR, vabs);
  k_scales<<<NE, 128, 0, stream>>>(e_feat, we1, be1, we2, be2, scales);
  k_final<<<(BB * NE) / 16, 128, 0, stream>>>(vabs, scales, pk_wo1, bo1,
                                              pk_wo2, bo2, pk_wo3, bo3,
                                              (float*)d_out);
}
